// MultiHeadedAttention_32607391711713
// MI455X (gfx1250) — compile-verified, matrix operand order checked by reading, not run on hardware
//
#include <hip/hip_runtime.h>
#include <hip/hip_bf16.h>

// ---- problem constants (from reference) ----
constexpr int Bc  = 2;
constexpr int Sc  = 2048;
constexpr int Hc  = 512;
constexpr int NHc = 8;
constexpr int DKc = 64;           // Hc / NHc
constexpr int Mrows = Bc * Sc;    // 4096 rows for projection GEMMs

typedef _Float16 f16_t;
typedef __attribute__((ext_vector_type(8)))  _Float16 v8h;
typedef __attribute__((ext_vector_type(16))) _Float16 v16h;
typedef __attribute__((ext_vector_type(8)))  float    v8f;

// D = A(16x32 f16) * B(32x16 f16) + C(16x16 f32)
__device__ __forceinline__ v8f wmma16x16x32(v16h a, v16h b, v8f c) {
    return __builtin_amdgcn_wmma_f32_16x16x32_f16(
        /*neg_a=*/false, a, /*neg_b=*/false, b,
        /*c_mod=*/(short)0, c, /*reuse_a=*/false, /*reuse_b=*/false);
}

// Load a 16x32 f16 fragment (A layout; B layout is the symmetric transpose,
// so the same loader serves B tiles when "row0" indexes the N dimension and
// the source is laid out [N][K] row-major).
// Per-lane: two contiguous 16-byte segments: K in [off,off+8) and [off+16,off+24),
// off = 8*(lane>=16), row = lane&15.  (CDNA5 ISA, 16-bit A 16x32)
__device__ __forceinline__ v16h load_frag(const f16_t* p, int ld, int row0, int k0) {
    const int lane = threadIdx.x & 31;
    const int grp  = lane >> 4;
    const int ln   = lane & 15;
    const f16_t* q = p + (size_t)(row0 + ln) * ld + k0 + grp * 8;
    union { v16h v; v8h h[2]; } u;
    u.h[0] = *(const v8h*)(q);
    u.h[1] = *(const v8h*)(q + 16);
    return u.v;
}

// ---- kernel 0: f32 -> f16 conversion (grid-stride) ----
__global__ void cvt_f32_to_f16(const float* __restrict__ src,
                               f16_t* __restrict__ dst, int n) {
    int i = blockIdx.x * blockDim.x + threadIdx.x;
    int stride = gridDim.x * blockDim.x;
    for (; i < n; i += stride) dst[i] = (f16_t)src[i];
}

// ---- kernel 1: fused Q/K/V projection: Y = X @ W^T + b ----
// blockIdx.z selects which projection; 8 waves per block, each wave owns a
// 16x64 output tile (one head's worth of columns).
__global__ __launch_bounds__(256)
void proj_kernel(const f16_t* __restrict__ Xq, const f16_t* __restrict__ Xk,
                 const f16_t* __restrict__ Xv,
                 const f16_t* __restrict__ Wq, const f16_t* __restrict__ Wk,
                 const f16_t* __restrict__ Wv,
                 const float* __restrict__ bq, const float* __restrict__ bk,
                 const float* __restrict__ bv,
                 f16_t* __restrict__ qh, f16_t* __restrict__ kh,
                 f16_t* __restrict__ vt) {
    const int which = blockIdx.z;
    const f16_t* X = (which == 0) ? Xq : (which == 1) ? Xk : Xv;
    const f16_t* W = (which == 0) ? Wq : (which == 1) ? Wk : Wv;
    const float* bb = (which == 0) ? bq : (which == 1) ? bk : bv;

    const int wid  = threadIdx.x >> 5;
    const int lane = threadIdx.x & 31;
    const int grp  = lane >> 4;
    const int ln   = lane & 15;
    const int row0 = blockIdx.x * 16;   // global M row
    const int n0   = wid * 64;          // 64 output cols == one head

    v8f acc[4] = {};
    for (int kk = 0; kk < Hc; kk += 32) {
        v16h a = load_frag(X, Hc, row0, kk);
#pragma unroll
        for (int t = 0; t < 4; ++t) {
            v16h bf = load_frag(W, Hc, n0 + t * 16, kk);   // rows of W == cols of W^T
            acc[t] = wmma16x16x32(a, bf, acc[t]);
        }
    }

    const int bidx = row0 >> 11;          // batch (S=2048 rows per batch)
    const int s0   = row0 & (Sc - 1);
    const float scale = (which == 0) ? 0.125f : 1.0f;   // 1/sqrt(DK) folded into Q

#pragma unroll
    for (int t = 0; t < 4; ++t) {
        const int col = n0 + t * 16 + ln;
        const float badd = bb[col];
        const int h  = col >> 6;          // == wid
        const int dk = col & 63;
        if (which < 2) {
            // Q/K: head-split row-major [bh][s][dk]
            f16_t* dst = ((which == 0) ? qh : kh) +
                         (size_t)(bidx * NHc + h) * Sc * DKc;
#pragma unroll
            for (int r = 0; r < 8; ++r) {
                int s = s0 + r + 8 * grp;
                dst[(size_t)s * DKc + dk] = (f16_t)((acc[t][r] + badd) * scale);
            }
        } else {
            // V: transposed [bh][dk][s]; 8 consecutive s -> one 16B store
            v8h pk;
#pragma unroll
            for (int r = 0; r < 8; ++r) pk[r] = (f16_t)(acc[t][r] + badd);
            f16_t* dst = vt + ((size_t)(bidx * NHc + h) * DKc + dk) * Sc
                            + s0 + 8 * grp;
            *(v8h*)dst = pk;
        }
    }
}

// ---- kernel 2: flash attention, one wave per 16-row q tile ----
__global__ __launch_bounds__(256)
void attn_kernel(const f16_t* __restrict__ qh, const f16_t* __restrict__ kh,
                 const f16_t* __restrict__ vt,
                 const float* __restrict__ bias, const int* __restrict__ mask,
                 float* __restrict__ out) {
    __shared__ f16_t lds[8][16 * 32];     // per-wave P bounce buffer (1 KB each)

    const int wid  = threadIdx.x >> 5;
    const int lane = threadIdx.x & 31;
    const int grp  = lane >> 4;
    const int ln   = lane & 15;

    const int gw    = blockIdx.x * 8 + wid;       // 2048 waves total
    const int qtile = gw & 127;                   // S/16 = 128 tiles
    const int bh    = gw >> 7;                    // 16 (b,h) pairs
    const int b     = bh >> 3;
    const int h     = bh & 7;
    const int q0    = qtile * 16;

    const f16_t* qb = qh + (size_t)bh * Sc * DKc;
    const f16_t* kb = kh + (size_t)bh * Sc * DKc;
    const f16_t* vb = vt + (size_t)bh * DKc * Sc;
    const float* biasb = bias + (size_t)b * Sc * Sc;
    const int*   maskb = mask + b * Sc;
    f16_t* pl = lds[wid];

    // Q fragments held in registers for the whole K loop (scale pre-folded)
    v16h qf0 = load_frag(qb, DKc, q0, 0);
    v16h qf1 = load_frag(qb, DKc, q0, 32);

    v8f acc[4] = {};
    float m[8], l[8];
#pragma unroll
    for (int r = 0; r < 8; ++r) { m[r] = -1e30f; l[r] = 0.0f; }

    for (int key0 = 0; key0 < Sc; key0 += 32) {
        // ---- scores: two 16x16 tiles (Q . K^T), K-dim 64 split into 2 WMMA
        v8f sc[2];
#pragma unroll
        for (int t = 0; t < 2; ++t) {
            v16h kf0 = load_frag(kb, DKc, key0 + t * 16, 0);
            v16h kf1 = load_frag(kb, DKc, key0 + t * 16, 32);
            v8f s = {};
            s = wmma16x16x32(qf0, kf0, s);
            s = wmma16x16x32(qf1, kf1, s);
            const int key = key0 + t * 16 + ln;
            const float madd = (float)maskb[key];   // reference adds 0/1 mask
#pragma unroll
            for (int r = 0; r < 8; ++r) {
                const int qrow = q0 + r + 8 * grp;
                const float bz = biasb[(size_t)qrow * Sc + key];
                float v = s[r] + madd;
                sc[t][r] = (bz == 0.0f) ? -1e30f : v;  // masked_fill on bias==0
            }
        }

        // ---- online softmax: rows live on 16-lane halves
        float alpha[8];
#pragma unroll
        for (int r = 0; r < 8; ++r) {
            float mx = fmaxf(sc[0][r], sc[1][r]);
#pragma unroll
            for (int off = 8; off >= 1; off >>= 1)
                mx = fmaxf(mx, __shfl_xor(mx, off, 32));
            const float mn = fmaxf(m[r], mx);
            alpha[r] = __expf(m[r] - mn);
            m[r] = mn;
            const float p0 = __expf(sc[0][r] - mn);
            const float p1 = __expf(sc[1][r] - mn);
            sc[0][r] = p0; sc[1][r] = p1;
            float ls = p0 + p1;
#pragma unroll
            for (int off = 8; off >= 1; off >>= 1)
                ls += __shfl_xor(ls, off, 32);
            l[r] = l[r] * alpha[r] + ls;
        }
#pragma unroll
        for (int t2 = 0; t2 < 4; ++t2)
#pragma unroll
            for (int r = 0; r < 8; ++r) acc[t2][r] *= alpha[r];

        // ---- repack P: C/D layout -> A layout via LDS
        __syncthreads();
#pragma unroll
        for (int t = 0; t < 2; ++t)
#pragma unroll
            for (int r = 0; r < 8; ++r)
                pl[(r + 8 * grp) * 32 + t * 16 + ln] = (f16_t)sc[t][r];
        __syncthreads();
        v16h pf = load_frag(pl, 32, 0, 0);

        // ---- acc += P (16x32) @ V (32x64); V transposed so B tiles are contiguous
#pragma unroll
        for (int t2 = 0; t2 < 4; ++t2) {
            v16h vf = load_frag(vb, Sc, t2 * 16, key0);
            acc[t2] = wmma16x16x32(pf, vf, acc[t2]);
        }
    }

    // ---- epilogue: divide by row sum, store f32 [B,S,H]
#pragma unroll
    for (int t2 = 0; t2 < 4; ++t2) {
        const int dk = t2 * 16 + ln;
#pragma unroll
        for (int r = 0; r < 8; ++r) {
            const int s = q0 + r + 8 * grp;
            out[((size_t)b * Sc + s) * Hc + h * DKc + dk] =
                acc[t2][r] * (1.0f / l[r]);
        }
    }
}

extern "C" void kernel_launch(void* const* d_in, const int* in_sizes, int n_in,
                              void* d_out, int out_size, void* d_ws, size_t ws_size,
                              hipStream_t stream) {
    const float* query = (const float*)d_in[0];
    const float* key   = (const float*)d_in[1];
    const float* value = (const float*)d_in[2];
    const float* bias  = (const float*)d_in[3];
    const int*   mask  = (const int*)d_in[4];
    const float* Wq    = (const float*)d_in[5];
    const float* bq    = (const float*)d_in[6];
    const float* Wk    = (const float*)d_in[7];
    const float* bk    = (const float*)d_in[8];
    const float* Wv    = (const float*)d_in[9];
    const float* bv    = (const float*)d_in[10];
    float* out = (float*)d_out;

    // workspace layout (f16 elements)
    constexpr size_t NX = (size_t)Mrows * Hc;   // 2,097,152
    constexpr size_t NW = (size_t)Hc * Hc;      //   262,144
    f16_t* xq = (f16_t*)d_ws;
    f16_t* xk = xq + NX;
    f16_t* xv = xk + NX;
    f16_t* wq = xv + NX;
    f16_t* wk = wq + NW;
    f16_t* wv = wk + NW;
    f16_t* qh = wv + NW;        // [B*NH][S][DK]
    f16_t* kh = qh + NX;
    f16_t* vt = kh + NX;        // [B*NH][DK][S]

    // 0) convert f32 inputs to f16
    cvt_f32_to_f16<<<2048, 256, 0, stream>>>(query, xq, (int)NX);
    cvt_f32_to_f16<<<2048, 256, 0, stream>>>(key,   xk, (int)NX);
    cvt_f32_to_f16<<<2048, 256, 0, stream>>>(value, xv, (int)NX);
    cvt_f32_to_f16<<<512,  256, 0, stream>>>(Wq,    wq, (int)NW);
    cvt_f32_to_f16<<<512,  256, 0, stream>>>(Wk,    wk, (int)NW);
    cvt_f32_to_f16<<<512,  256, 0, stream>>>(Wv,    wv, (int)NW);

    // 1) projections (z = which of q/k/v)
    proj_kernel<<<dim3(Mrows / 16, 1, 3), 256, 0, stream>>>(
        xq, xk, xv, wq, wk, wv, bq, bk, bv, qh, kh, vt);

    // 2) flash attention
    attn_kernel<<<256, 256, 0, stream>>>(qh, kh, vt, bias, mask, out);
}
